// DynamicAdj_21577915695603
// MI455X (gfx1250) — matrix operand order repaired and checked by reading, not compiled, not run on hardware
//
#include <hip/hip_runtime.h>
#include <hip/hip_bf16.h>

// ---------------------------------------------------------------------------
// DynamicAdj network on MI455X (gfx1250), f16 WMMA implicit-GEMM convolutions.
// Activations: f16 NHWC [B,H,W,64]. Weights packed f16 [o][k] with
// k = (ky*3+kx)*64 + ci (so a B-matrix column is K-contiguous).
// ---------------------------------------------------------------------------

typedef __attribute__((ext_vector_type(16))) _Float16 v16h;
typedef __attribute__((ext_vector_type(8)))  _Float16 v8h;
typedef __attribute__((ext_vector_type(8)))  float    v8f;

#define NF 64
#define KTOT 576                 // 9 taps * 64 channels
#define WPACK_ELEMS (NF * KTOT)  // 36864 halves per conv layer
#define MTILES 4                 // M-tiles per wave per block (weight reuse)

// Optional gfx1250 async global->LDS staging (graceful fallback if missing).
#if defined(__has_builtin)
# if __has_builtin(__builtin_amdgcn_global_load_async_to_lds_b128) && \
     __has_builtin(__builtin_amdgcn_s_wait_asynccnt)
#  define USE_ASYNC_WLDS 1
# endif
#endif
#ifndef USE_ASYNC_WLDS
# define USE_ASYNC_WLDS 0
#endif

// Workspace layout (bytes). Requires ws_size >= ~189 MB.
#define OFF_X   ((size_t)0)                 // 64 MB  f16 [32,128,128,64]
#define OFF_T   ((size_t)(64u  << 20))      // 64 MB  f16 temp @128^2
#define OFF_Y   ((size_t)(128u << 20))      // 16 MB  f16 [32,64,64,64]
#define OFF_U   ((size_t)(144u << 20))      // 16 MB  f16 temp @64^2
#define OFF_T2  ((size_t)(160u << 20))      // 16 MB  f16 temp @64^2
#define OFF_FI  ((size_t)(176u << 20))      //  4 MB  f16 [32,32,32,64] intra
#define OFF_FE  ((size_t)(180u << 20))      //  4 MB  f16 inter
#define OFF_WP  ((size_t)(184u << 20))      // ~1.1MB f16 packed conv weights x15
#define OFF_F   ((size_t)(186u << 20))      // 512 KB fp32 head features [4][32][1024]
#define OFF_H   ((size_t)(187u << 20))      // 248 KB fp32 head outputs  [4][32][496]

// ---------------------------------------------------------------------------
// Repack conv3x3 weights: fp32 OIHW [64][64][3][3] -> f16 [o][ (tap)*64 + ci ]
// ---------------------------------------------------------------------------
__global__ __launch_bounds__(256)
void pack_w_k(const float* __restrict__ src, _Float16* __restrict__ dst) {
  int t = blockIdx.x * 256 + threadIdx.x;
  if (t >= WPACK_ELEMS) return;
  int o = t / KTOT, k = t % KTOT;
  int tap = k >> 6, ci = k & 63;
  dst[t] = (_Float16)src[(o * 64 + ci) * 9 + tap];
}

// ---------------------------------------------------------------------------
// First conv: 1x1, 3 -> 64, fp32 NCHW in, f16 NHWC out. One thread per pixel.
// ---------------------------------------------------------------------------
__global__ __launch_bounds__(256)
void first_conv_k(const float* __restrict__ x, const float* __restrict__ w,
                  const float* __restrict__ bias, _Float16* __restrict__ out) {
  const int pix = blockIdx.x * 256 + threadIdx.x;     // b*16384 + h*128 + w
  const int b  = pix >> 14;
  const int hw = pix & 16383;
  const float x0 = x[((long)b * 3 + 0) * 16384 + hw];
  const float x1 = x[((long)b * 3 + 1) * 16384 + hw];
  const float x2 = x[((long)b * 3 + 2) * 16384 + hw];
  _Float16 vals[NF];
#pragma unroll
  for (int o = 0; o < NF; ++o) {
    float v = bias[o] + w[o * 3 + 0] * x0 + w[o * 3 + 1] * x1 + w[o * 3 + 2] * x2;
    vals[o] = (_Float16)v;
  }
  uint4* dst = (uint4*)(out + (long)pix * NF);
  const uint4* s = (const uint4*)vals;
#pragma unroll
  for (int i = 0; i < 8; ++i) dst[i] = s[i];
}

// ---------------------------------------------------------------------------
// Inner K-loop for the implicit-GEMM conv. CHECK=0 is the branch-free interior
// fast path (wave-uniform guarantee that all taps are in-bounds).
// ---------------------------------------------------------------------------
template<int STRIDE, int RELU_IN, int CHECK>
__device__ __forceinline__ void conv_kloop(
    const _Float16* __restrict__ in, const _Float16* wlds,
    int b, int oh, int ow, int Hin, int Win,
    int kba, int kbb, int mn, v8f (&acc)[4]) {
#pragma unroll
  for (int tap = 0; tap < 9; ++tap) {
    const int ky = tap / 3, kx = tap % 3;
    const int ih = oh * STRIDE + ky - 1;
    const int iw = ow * STRIDE + kx - 1;
    bool inb = true;
    if (CHECK)
      inb = ((unsigned)ih < (unsigned)Hin) && ((unsigned)iw < (unsigned)Win);
    const _Float16* src = in + (((long)b * Hin + ih) * Win + iw) * NF;
#pragma unroll
    for (int kc = 0; kc < 2; ++kc) {       // two K=32 chunks of the 64-ch tap
      const int c0 = kc * 32;
      v16h a = {};
      if (inb) {
        // A fragment, contiguous K order (matches B): halves 0..15 -> K=kbb+i.
        // Two aligned 16B loads.
        v8h lo = *(const v8h*)(src + c0 + kbb);
        v8h hi = *(const v8h*)(src + c0 + kbb + 8);
#pragma unroll
        for (int i = 0; i < 8; ++i) { a[i] = lo[i]; a[8 + i] = hi[i]; }
      }
      if (RELU_IN) {
#pragma unroll
        for (int i = 0; i < 16; ++i)
          a[i] = (a[i] > (_Float16)0) ? a[i] : (_Float16)0;
      }
      const int kidx = tap * 64 + c0 + kbb;
#pragma unroll
      for (int nt = 0; nt < 4; ++nt) {
        // B fragment: column (nt*16+mn), K = kidx..kidx+15 contiguous in LDS.
        const v16h bf = *(const v16h*)(&wlds[(nt * 16 + mn) * KTOT + kidx]);
        acc[nt] = __builtin_amdgcn_wmma_f32_16x16x32_f16(
            false, a, false, bf, (short)0, acc[nt], false, false);
      }
    }
  }
}

// ---------------------------------------------------------------------------
// 3x3 conv, 64->64, implicit GEMM with v_wmma_f32_16x16x32_f16.
// Block = 256 threads = 8 waves. Weights staged once in LDS (72 KB, async
// global->LDS when available), then each wave processes MTILES=4 tiles of
// 16 pixels x 64 channels, amortizing the staging 4x.
// Grid: (B*Hout*Wout)/512 blocks (exact).
// ---------------------------------------------------------------------------
template<int STRIDE, int RELU_IN, int RELU_OUT, int RESIDUAL>
__global__ __launch_bounds__(256, 1)
void conv3x3_k(const _Float16* __restrict__ in, const _Float16* __restrict__ wp,
               const float* __restrict__ bias, const _Float16* __restrict__ res,
               _Float16* __restrict__ out, int Hin, int Win, int Hout, int Wout) {
  __shared__ _Float16 wlds[WPACK_ELEMS];       // 72 KB weights
  __shared__ _Float16 obuf[8][16 * NF];        // 16 KB per-wave output staging
#if USE_ASYNC_WLDS
  {
    // Builtin signature (from hipcc diagnostics): param0 = AS(1) int4*,
    // param1 = AS(3) int4*, then two imm ints (offset, cpol).
    typedef int v4si __attribute__((vector_size(16)));
    typedef __attribute__((address_space(1))) v4si as1_v4si;
    typedef __attribute__((address_space(3))) v4si as3_v4si;
#pragma unroll
    for (int i = 0; i < (WPACK_ELEMS * 2) / 16 / 256; ++i) {
      as1_v4si* gp = (as1_v4si*)((const uint4*)wp + i * 256 + threadIdx.x);
      as3_v4si* lp = (as3_v4si*)((uint4*)wlds + i * 256 + threadIdx.x);
      __builtin_amdgcn_global_load_async_to_lds_b128(gp, lp, 0, 0);
    }
    __builtin_amdgcn_s_wait_asynccnt(0);
  }
#else
  {
    const uint4* s = (const uint4*)wp;
    uint4* d = (uint4*)wlds;
#pragma unroll
    for (int i = 0; i < (WPACK_ELEMS * 2) / 16 / 256; ++i)
      d[i * 256 + threadIdx.x] = s[i * 256 + threadIdx.x];
  }
#endif
  __syncthreads();

  const int wave   = threadIdx.x >> 5;
  const int lane   = threadIdx.x & 31;
  const int half16 = lane >> 4;            // 0 (lanes 0-15) / 1 (lanes 16-31)
  const int mn     = lane & 15;            // A row (M) and B col (N) index
  const int HWo    = Hout * Wout;
  const int kba = half16 * 8;              // unused; A loads use kbb
  const int kbb = half16 * 16;             // A and B: K base, contiguous K order

#pragma unroll 1
  for (int mt = 0; mt < MTILES; ++mt) {
    const long mbase = (long)blockIdx.x * (128 * MTILES) + (mt * 8 + wave) * 16;
    // Speculative prefetch of the next tile's input row (stride-1 layers).
    if (STRIDE == 1 && mt + 1 < MTILES) {
      const long nb = (long)blockIdx.x * (128 * MTILES) + ((mt + 1) * 8 + wave) * 16 + mn;
      __builtin_prefetch((const void*)(in + nb * NF), 0, 0);
    }
    // Tiles are 16-aligned and Wout % 16 == 0 -> a tile never crosses a row,
    // so (b, oh, ow0) are wave-uniform.
    const int b   = (int)(mbase / HWo);
    const int hw0 = (int)(mbase % HWo);
    const int oh  = hw0 / Wout;
    const int ow0 = hw0 % Wout;

    v8f acc[4] = {};                       // 4 N-tiles of 16 channels

    const bool interior =
        (oh * STRIDE - 1 >= 0) && (oh * STRIDE + 1 <= Hin - 1) &&
        (ow0 * STRIDE - 1 >= 0) && ((ow0 + 15) * STRIDE + 1 <= Win - 1);

    if (interior)   // wave-uniform branch: branch-free inner loop
      conv_kloop<STRIDE, RELU_IN, 0>(in, wlds, b, oh, ow0 + mn, Hin, Win,
                                     kba, kbb, mn, acc);
    else
      conv_kloop<STRIDE, RELU_IN, 1>(in, wlds, b, oh, ow0 + mn, Hin, Win,
                                     kba, kbb, mn, acc);

    // Epilogue: C/D layout -> lane holds N = nt*16+mn, rows M = half16*8 + j.
    // Stage through wave-private LDS slice, then coalesced b128 stores.
    // LDS ops are in-order within a wave: safe to reuse obuf across mt without
    // a barrier, and no cross-wave sharing.
    _Float16* ob = &obuf[wave][0];
    const int rbase = half16 * 8;
#pragma unroll
    for (int nt = 0; nt < 4; ++nt) {
      const int nc = nt * 16 + mn;
      const float bs = bias[nc];
#pragma unroll
      for (int j = 0; j < 8; ++j) {
        float v = acc[nt][j] + bs;
        if (RELU_OUT) v = (v > 0.f) ? v : 0.f;
        ob[(rbase + j) * NF + nc] = (_Float16)v;
      }
    }
    const int p  = lane >> 1;              // pixel within tile
    const int cs = (lane & 1) * 32;        // channel chunk (32 ch = 64 B)
    const long pm = mbase + p;
    const _Float16* lsrc = ob + p * NF + cs;
    _Float16* gdst = out + pm * NF + cs;
    if (RESIDUAL) {
      const _Float16* rsrc = res + pm * NF + cs;
#pragma unroll
      for (int q = 0; q < 4; ++q) {
        v8h vv = *(const v8h*)(lsrc + q * 8);
        v8h rr = *(const v8h*)(rsrc + q * 8);
#pragma unroll
        for (int i = 0; i < 8; ++i) vv[i] = vv[i] + rr[i];
        *(v8h*)(gdst + q * 8) = vv;
      }
    } else {
#pragma unroll
      for (int q = 0; q < 4; ++q)
        *(v8h*)(gdst + q * 8) = *(const v8h*)(lsrc + q * 8);
    }
  }
}

// ---------------------------------------------------------------------------
// Head features: f[head][b][p] = bias + dot64(feat[b,p,:], w).  131072 threads.
// heads 0,1 read intra features; 2,3 read inter features.
// ---------------------------------------------------------------------------
__global__ __launch_bounds__(256)
void head_feat_k(const _Float16* __restrict__ fi, const _Float16* __restrict__ fe,
                 const float* __restrict__ w0, const float* __restrict__ b0,
                 const float* __restrict__ w1, const float* __restrict__ b1,
                 const float* __restrict__ w2, const float* __restrict__ b2,
                 const float* __restrict__ w3, const float* __restrict__ b3,
                 float* __restrict__ f) {
  const int t = blockIdx.x * 256 + threadIdx.x;   // 4*32*1024
  const int head = t >> 15;
  const int bp   = t & 32767;
  const _Float16* feat = ((head < 2) ? fi : fe) + (long)bp * NF;
  const float* w  = (head == 0) ? w0 : (head == 1) ? w1 : (head == 2) ? w2 : w3;
  const float* bb = (head == 0) ? b0 : (head == 1) ? b1 : (head == 2) ? b2 : b3;
  float acc = bb[0];
#pragma unroll
  for (int c = 0; c < NF; ++c) acc += (float)feat[c] * w[c];
  f[t] = acc;
}

// ---------------------------------------------------------------------------
// Head linears + clip: h[head][b][o] = (f[head][b] . W[o] + bias[o]) > 0.
// 4*32*496 = 63488 threads (496 blocks x 128).
// ---------------------------------------------------------------------------
__global__ __launch_bounds__(128)
void head_lin_k(const float* __restrict__ f,
                const float* __restrict__ W0, const float* __restrict__ c0,
                const float* __restrict__ W1, const float* __restrict__ c1,
                const float* __restrict__ W2, const float* __restrict__ c2,
                const float* __restrict__ W3, const float* __restrict__ c3,
                float* __restrict__ h) {
  const int t = blockIdx.x * 128 + threadIdx.x;
  const int head = t / (32 * 496);
  const int r = t % (32 * 496);
  const int b = r / 496, o = r % 496;
  const float* W  = (head == 0) ? W0 : (head == 1) ? W1 : (head == 2) ? W2 : W3;
  const float* bb = (head == 0) ? c0 : (head == 1) ? c1 : (head == 2) ? c2 : c3;
  const float* fv = f + ((long)head * 32 + b) * 1024;
  const float* wr = W + (long)o * 1024;
  float acc = bb[o];
#pragma unroll 8
  for (int p = 0; p < 1024; ++p) acc += fv[p] * wr[p];
  h[t] = (acc > 0.f) ? 1.f : 0.f;
}

// ---------------------------------------------------------------------------
// Build 32x32 adjacency from 496 upper-tri values, symmetrize + I, and emit
// both the rsqrt-degree-normalized and raw matrices. One block per (slot,b).
// out: [8][32][32][32] fp32; slots 0..3 = normalized, 4..7 = raw.
// ---------------------------------------------------------------------------
__global__ __launch_bounds__(1024)
void adj_k(const float* __restrict__ h, float* __restrict__ out) {
  __shared__ float A[1024];
  __shared__ float d[32];
  const int s = blockIdx.x >> 5;
  const int b = blockIdx.x & 31;
  const int t = threadIdx.x;
  const int i = t >> 5, j = t & 31;
  const float* hv = h + ((long)s * 32 + b) * 496;
  float v;
  if (i == j) {
    v = 1.f;
  } else {
    const int a = (i < j) ? i : j;
    const int c = (i < j) ? j : i;
    const int idx = a * (63 - a) / 2 + (c - a - 1);   // triu_indices(32,1) row-major
    v = hv[idx];
  }
  A[t] = v;
  __syncthreads();
  if (t < 32) {
    float sum = 0.f;
#pragma unroll
    for (int k = 0; k < 32; ++k) sum += A[t * 32 + k];
    d[t] = rsqrtf(sum);
  }
  __syncthreads();
  out[((long)(4 + s) * 32 + b) * 1024 + t] = v;
  out[((long)s * 32 + b) * 1024 + t] = v * d[i] * d[j];
}

// ---------------------------------------------------------------------------
// Launcher
// ---------------------------------------------------------------------------
extern "C" void kernel_launch(void* const* d_in, const int* in_sizes, int n_in,
                              void* d_out, int out_size, void* d_ws, size_t ws_size,
                              hipStream_t stream) {
  (void)in_sizes; (void)n_in; (void)out_size; (void)ws_size;
  // d_in order (setup_inputs insertion order, tuples flattened (w,b)):
  // 0:x  1,2:first  3..10:share_body(4x w,b)  11,12:share_down
  // 13..20:intra_body  21,22:intra_down  23..30:inter_body  31,32:inter_down
  // 33..40:{intra_v,intra_h,inter_v,inter_h}_head  41..48:{..}_body linears
  char* ws = (char*)d_ws;
  _Float16* X  = (_Float16*)(ws + OFF_X);
  _Float16* T  = (_Float16*)(ws + OFF_T);
  _Float16* Y  = (_Float16*)(ws + OFF_Y);
  _Float16* U  = (_Float16*)(ws + OFF_U);
  _Float16* T2 = (_Float16*)(ws + OFF_T2);
  _Float16* FI = (_Float16*)(ws + OFF_FI);
  _Float16* FE = (_Float16*)(ws + OFF_FE);
  _Float16* WP = (_Float16*)(ws + OFF_WP);
  float*    F  = (float*)(ws + OFF_F);
  float*    H  = (float*)(ws + OFF_H);
  float*    out = (float*)d_out;

  const int widx[15] = {3,5,7,9, 11, 13,15,17,19, 21, 23,25,27,29, 31};
  for (int l = 0; l < 15; ++l)
    pack_w_k<<<(WPACK_ELEMS + 255) / 256, 256, 0, stream>>>(
        (const float*)d_in[widx[l]], WP + (size_t)l * WPACK_ELEMS);

  first_conv_k<<<2048, 256, 0, stream>>>(
      (const float*)d_in[0], (const float*)d_in[1], (const float*)d_in[2], X);

  #define WPL(l) (WP + (size_t)(l) * WPACK_ELEMS)
  #define BIA(i) ((const float*)d_in[(i)])

  // ---- shared trunk @128^2 (524288 px / 512 = 1024 blocks) ----
  conv3x3_k<1,0,1,0><<<1024,256,0,stream>>>(X, WPL(0),  BIA(4),  nullptr, T, 128,128,128,128);
  conv3x3_k<1,0,0,1><<<1024,256,0,stream>>>(T, WPL(1),  BIA(6),  X,       X, 128,128,128,128);
  conv3x3_k<1,0,1,0><<<1024,256,0,stream>>>(X, WPL(2),  BIA(8),  nullptr, T, 128,128,128,128);
  conv3x3_k<1,0,0,1><<<1024,256,0,stream>>>(T, WPL(3),  BIA(10), X,       X, 128,128,128,128);
  conv3x3_k<2,0,0,0><<< 256,256,0,stream>>>(X, WPL(4),  BIA(12), nullptr, Y, 128,128, 64, 64);

  // ---- intra branch @64^2 (131072 px / 512 = 256 blocks) ----
  conv3x3_k<1,0,1,0><<< 256,256,0,stream>>>(Y,  WPL(5),  BIA(14), nullptr, T2, 64,64,64,64);
  conv3x3_k<1,0,0,1><<< 256,256,0,stream>>>(T2, WPL(6),  BIA(16), Y,       U,  64,64,64,64);
  conv3x3_k<1,0,1,0><<< 256,256,0,stream>>>(U,  WPL(7),  BIA(18), nullptr, T2, 64,64,64,64);
  conv3x3_k<1,0,0,1><<< 256,256,0,stream>>>(T2, WPL(8),  BIA(20), U,       U,  64,64,64,64);
  conv3x3_k<2,1,0,0><<<  64,256,0,stream>>>(U,  WPL(9),  BIA(22), nullptr, FI, 64,64,32,32);

  // ---- inter branch @64^2 ----
  conv3x3_k<1,0,1,0><<< 256,256,0,stream>>>(Y,  WPL(10), BIA(24), nullptr, T2, 64,64,64,64);
  conv3x3_k<1,0,0,1><<< 256,256,0,stream>>>(T2, WPL(11), BIA(26), Y,       U,  64,64,64,64);
  conv3x3_k<1,0,1,0><<< 256,256,0,stream>>>(U,  WPL(12), BIA(28), nullptr, T2, 64,64,64,64);
  conv3x3_k<1,0,0,1><<< 256,256,0,stream>>>(T2, WPL(13), BIA(30), U,       U,  64,64,64,64);
  conv3x3_k<2,1,0,0><<<  64,256,0,stream>>>(U,  WPL(14), BIA(32), nullptr, FE, 64,64,32,32);

  // ---- heads ----
  head_feat_k<<<512, 256, 0, stream>>>(FI, FE,
      BIA(33), BIA(34), BIA(35), BIA(36), BIA(37), BIA(38), BIA(39), BIA(40), F);
  head_lin_k<<<496, 128, 0, stream>>>(F,
      BIA(41), BIA(42), BIA(43), BIA(44), BIA(45), BIA(46), BIA(47), BIA(48), H);

  // ---- adjacency build + normalize ----
  adj_k<<<128, 1024, 0, stream>>>(H, out);

  #undef WPL
  #undef BIA
}
